// DeepseekMoE_48455821033513
// MI455X (gfx1250) — compile-verified, matrix operand order checked by reading, not run on hardware
//
#include <hip/hip_runtime.h>
#include <hip/hip_bf16.h>

namespace {

constexpr int   kT    = 1024;
constexpr int   kH    = 1024;
constexpr int   kE    = 64;
constexpr int   kK    = 6;
constexpr int   kTG   = 4;
constexpr int   kF    = 512;
constexpr int   kFS   = 1024;   // F * NSH
constexpr int   kCAP  = 384;
constexpr float kSCALE = 2.5f;

typedef __attribute__((ext_vector_type(16))) __bf16 v16bf;
typedef __attribute__((ext_vector_type(8)))  __bf16 v8bf;
typedef __attribute__((ext_vector_type(4)))  __bf16 v4bf;
typedef __attribute__((ext_vector_type(2)))  __bf16 v2bf;
typedef __attribute__((ext_vector_type(8)))  float  v8f;
typedef __attribute__((ext_vector_type(4)))  float  v4f;
typedef unsigned int u32x4 __attribute__((ext_vector_type(4)));
typedef int          i32x4 __attribute__((ext_vector_type(4)));
typedef int          i32x8 __attribute__((ext_vector_type(8)));

// Build a 16-wide bf16 WMMA fragment from an LDS row laid out [idx][k0..k31].
// Elements 0..7  <- K = khalf*8 + 0..7
// Elements 8..15 <- K = 16 + khalf*8 + 0..7   (khalf = lane>>4)
__device__ __forceinline__ v16bf ldsFrag(const __bf16* base, int khalf) {
  v16bf f;
  ((v8bf*)&f)[0] = *(const v8bf*)(base + (khalf << 3));
  ((v8bf*)&f)[1] = *(const v8bf*)(base + 16 + (khalf << 3));
  return f;
}

__device__ __forceinline__ float siluf(float x) {
  return x / (1.0f + __expf(-x));
}

// TDM: async 2D tile copy global->LDS (bf16 payload, data_size = 2 bytes).
// Descriptor per CDNA5 ISA D# group0/group1 layout.
__device__ __forceinline__ void tdm_load_2d_bf16(unsigned lds_off,
                                                 const void* gptr,
                                                 int rows, int cols,
                                                 int stride_elems) {
  unsigned long long ga = (unsigned long long)(uintptr_t)gptr;
  u32x4 g0;
  g0[0] = 1u;                                    // count=1, user descriptor
  g0[1] = lds_off;                               // lds_addr (bytes)
  g0[2] = (unsigned)(ga & 0xffffffffull);        // global_addr[31:0]
  g0[3] = (unsigned)((ga >> 32) & 0x01ffffffull) | (2u << 30);  // [56:32]|type=2
  i32x8 g1;
  g1[0] = (1 << 16);                             // wg_mask=0, data_size=1 (2B)
  g1[1] = (cols & 0xffff) << 16;                 // tensor_dim0 [15:0] @ bits 63:48
  g1[2] = ((unsigned)cols >> 16) | ((rows & 0xffff) << 16);  // dim0 hi | dim1 lo
  g1[3] = (((unsigned)rows >> 16) & 0xffff) | ((cols & 0xffff) << 16); // |tile_dim0
  g1[4] = (rows & 0xffff);                       // tile_dim1, tile_dim2=0
  g1[5] = stride_elems;                          // tensor_dim0_stride lo32
  g1[6] = 0;
  g1[7] = 0;
  i32x4 z4 = {0, 0, 0, 0};
#if __has_include(<hip/amd_detail/amd_gfx1250_TDM.h>)
  i32x8 z8 = {0, 0, 0, 0, 0, 0, 0, 0};
  __builtin_amdgcn_tensor_load_to_lds(g0, g1, z4, z4, z8, 0);
#else
  __builtin_amdgcn_tensor_load_to_lds(g0, g1, z4, z4, 0);
#endif
}

// ---------------------------------------------------------------------------
// 0) workspace init: counters + slot tables
// ---------------------------------------------------------------------------
__global__ void moe_init_kernel(int* __restrict__ ecount,
                                int* __restrict__ slot_token,
                                float* __restrict__ slot_weight) {
  int i = blockIdx.x * blockDim.x + threadIdx.x;
  if (i < kE) ecount[i] = 0;
  if (i < kE * kCAP) { slot_token[i] = 0; slot_weight[i] = 0.0f; }
}

// ---------------------------------------------------------------------------
// 1) gate logits + grouped top-k routing; one wave32 per token
// ---------------------------------------------------------------------------
__global__ void gate_topk_kernel(const float* __restrict__ X,
                                 const float* __restrict__ GW,
                                 const float* __restrict__ BIAS,
                                 int* __restrict__ ecount,
                                 int* __restrict__ slot_token,
                                 float* __restrict__ slot_weight) {
  const int t = blockIdx.x;
  const int lane = threadIdx.x;  // 0..31
  const float* xr = X + (size_t)t * kH;
  const float* g0 = GW + (size_t)lane * kH;
  const float* g1 = GW + (size_t)(lane + 32) * kH;
  float a0 = 0.0f, a1 = 0.0f;
  for (int h = 0; h < kH; h += 4) {
    v4f xv = *(const v4f*)(xr + h);
    v4f w0 = *(const v4f*)(g0 + h);
    v4f w1 = *(const v4f*)(g1 + h);
    for (int j = 0; j < 4; ++j) { a0 += xv[j] * w0[j]; a1 += xv[j] * w1[j]; }
  }
  __shared__ float sc[kE];  // sigmoid scores
  __shared__ float ss[kE];  // scores + bias
  float s0 = 1.0f / (1.0f + __expf(-a0));
  float s1 = 1.0f / (1.0f + __expf(-a1));
  sc[lane] = s0;        ss[lane] = s0 + BIAS[lane];
  sc[lane + 32] = s1;   ss[lane + 32] = s1 + BIAS[lane + 32];
  __syncthreads();
  if (lane == 0) {
    float gsc[8];
    for (int g = 0; g < 8; ++g) {
      float m1 = -1e30f, m2 = -1e30f;
      for (int i = 0; i < 8; ++i) {
        float v = ss[g * 8 + i];
        if (v > m1) { m2 = m1; m1 = v; } else if (v > m2) { m2 = v; }
      }
      gsc[g] = m1 + m2;
    }
    bool gsel[8] = {false, false, false, false, false, false, false, false};
    for (int s = 0; s < kTG; ++s) {
      int bi = 0; float bv = -1e30f;
      for (int g = 0; g < 8; ++g)
        if (!gsel[g] && gsc[g] > bv) { bv = gsc[g]; bi = g; }
      gsel[bi] = true;
    }
    float msk[kE];
    for (int e = 0; e < kE; ++e) msk[e] = gsel[e >> 3] ? ss[e] : -1e30f;
    int ids[kK]; float wv[kK]; float wsum = 0.0f;
    for (int s = 0; s < kK; ++s) {
      int bi = 0; float bv = -1e30f;
      for (int e = 0; e < kE; ++e)
        if (msk[e] > bv) { bv = msk[e]; bi = e; }
      msk[bi] = -1e30f;
      ids[s] = bi;
      wv[s] = sc[bi];      // weights from raw sigmoid scores
      wsum += wv[s];
    }
    float scl = kSCALE / wsum;
    for (int s = 0; s < kK; ++s) {
      int ee = ids[s];
      int slot = atomicAdd(&ecount[ee], 1);
      if (slot < kCAP) {
        slot_token[ee * kCAP + slot] = t;
        slot_weight[ee * kCAP + slot] = wv[s] * scl;
      }
    }
  }
}

// ---------------------------------------------------------------------------
// 2) gate/up GEMM + SiLU*mul -> bf16 activation
//    Block: 128 threads (4 waves); block tile = 32 rows x (64 gate + 64 up).
//    Per wave: 2 row sub-tiles x (gate + up) = 4 WMMA per K-step; B fragments
//    are each reused by two WMMAs.
// ---------------------------------------------------------------------------
template <bool GATHER>
__global__ void moe_gateup_kernel(const float* __restrict__ X,
                                  const float* __restrict__ B,
                                  const int* __restrict__ slot_token,
                                  const int* __restrict__ ecount,
                                  __bf16* __restrict__ ACT,
                                  int Fd) {  // 512 routed / 1024 shared
  const int tid  = threadIdx.x;
  const int lane = tid & 31;
  const int wv   = tid >> 5;
  const int cg    = blockIdx.x;       // 64-col group
  const int rbase = blockIdx.y * 32;  // 32-row tile
  const int e     = GATHER ? blockIdx.z : 0;
  const int bstride = 2 * Fd;

  if (GATHER) {
    int cnt = ecount[e]; if (cnt > kCAP) cnt = kCAP;
    if (rbase >= cnt) return;
  }
  const float* Bp = B + (GATHER ? (size_t)e * kH * bstride : (size_t)0);

  __shared__ int    s_tok[32];
  __shared__ __bf16 sA[32 * 32];    // [row][k]
  __shared__ __bf16 sB[128 * 32];   // [col][k], cols 0..63 gate, 64..127 up

  if (tid < 32) {
    s_tok[tid] = GATHER ? slot_token[e * kCAP + rbase + tid] : (rbase + tid);
  }
  __syncthreads();

  v8f accG0 = {0.f,0.f,0.f,0.f,0.f,0.f,0.f,0.f};
  v8f accG1 = {0.f,0.f,0.f,0.f,0.f,0.f,0.f,0.f};
  v8f accU0 = {0.f,0.f,0.f,0.f,0.f,0.f,0.f,0.f};
  v8f accU1 = {0.f,0.f,0.f,0.f,0.f,0.f,0.f,0.f};

  const int khalf = lane >> 4;
  const int mrow  = lane & 15;

  for (int k0 = 0; k0 < kH; k0 += 32) {
    // ---- stage A: 32 rows x 32 k, fp32 -> bf16, gathered rows ----
    {
      int row = tid >> 2;
      int c0  = (tid & 3) << 3;
      const float* xp = X + (size_t)s_tok[row] * kH + k0 + c0;
      v4f xa = *(const v4f*)xp;
      v4f xb2 = *(const v4f*)(xp + 4);
      v8bf xb;
      for (int j = 0; j < 4; ++j) { xb[j] = (__bf16)xa[j]; xb[4+j] = (__bf16)xb2[j]; }
      *(v8bf*)&sA[row * 32 + c0] = xb;
    }
    // ---- stage B transposed: 128 cols x 32 k ----
    for (int it = 0; it < 2; ++it) {
      int ch = tid + (it << 7);       // 0..255
      int kp = ch & 15;               // k-pair 0..15
      int cc = ch >> 4;               // col chunk 0..15
      int lcol0 = cc << 3;
      int gcol0 = (lcol0 < 64) ? (cg * 64 + lcol0)
                               : (Fd + cg * 64 + (lcol0 - 64));
      const float* bp = Bp + (size_t)(k0 + kp * 2) * bstride + gcol0;
      v4f r0a = *(const v4f*)bp;
      v4f r0b = *(const v4f*)(bp + 4);
      v4f r1a = *(const v4f*)(bp + bstride);
      v4f r1b = *(const v4f*)(bp + bstride + 4);
      for (int j = 0; j < 4; ++j) {
        v2bf p0; p0[0] = (__bf16)r0a[j]; p0[1] = (__bf16)r1a[j];
        *(v2bf*)&sB[(lcol0 + j) * 32 + kp * 2] = p0;
        v2bf p1; p1[0] = (__bf16)r0b[j]; p1[1] = (__bf16)r1b[j];
        *(v2bf*)&sB[(lcol0 + 4 + j) * 32 + kp * 2] = p1;
      }
      if (k0 + 32 < kH)
        __builtin_prefetch(bp + (size_t)32 * bstride, 0, 3);  // next K-tile
    }
    __syncthreads();

    // ---- compute ----
    v16bf af0 = ldsFrag(&sA[mrow * 32], khalf);
    v16bf af1 = ldsFrag(&sA[(16 + mrow) * 32], khalf);
    v16bf bg  = ldsFrag(&sB[(wv * 16 + mrow) * 32], khalf);
    v16bf bu  = ldsFrag(&sB[(64 + wv * 16 + mrow) * 32], khalf);
    accG0 = __builtin_amdgcn_wmma_f32_16x16x32_bf16(false, af0, false, bg,
                                                    (short)0, accG0, false, false);
    accG1 = __builtin_amdgcn_wmma_f32_16x16x32_bf16(false, af1, false, bg,
                                                    (short)0, accG1, false, false);
    accU0 = __builtin_amdgcn_wmma_f32_16x16x32_bf16(false, af0, false, bu,
                                                    (short)0, accU0, false, false);
    accU1 = __builtin_amdgcn_wmma_f32_16x16x32_bf16(false, af1, false, bu,
                                                    (short)0, accU1, false, false);
    __syncthreads();
  }

  // ---- epilogue: SiLU(gate) * up -> bf16 activation ----
  const int n   = lane & 15;
  const int mhi = (lane >> 4) << 3;
  const int col = cg * 64 + wv * 16 + n;
  const size_t rowbase = GATHER ? ((size_t)e * kCAP + rbase) : (size_t)rbase;
  for (int r = 0; r < 8; ++r) {
    int m = r + mhi;
    ACT[(rowbase + m) * Fd + col]      = (__bf16)(siluf(accG0[r]) * accU0[r]);
    ACT[(rowbase + 16 + m) * Fd + col] = (__bf16)(siluf(accG1[r]) * accU1[r]);
  }
}

// ---------------------------------------------------------------------------
// 3) down GEMM. A (bf16 activations) staged via TDM tensor_load_to_lds;
//    B (fp32 weights) converted in flight. Block tile = 32 rows x 64 cols.
//    ROUTED=true : act[e] @ w2[e], scatter-add into out with routing weight
//    ROUTED=false: act_s @ sw2, plain store (initializes full output)
// ---------------------------------------------------------------------------
template <bool ROUTED>
__global__ void moe_down_kernel(const __bf16* __restrict__ ACT,
                                const float* __restrict__ B,
                                const int* __restrict__ slot_token,
                                const float* __restrict__ slot_weight,
                                const int* __restrict__ ecount,
                                float* __restrict__ OUT,
                                int Kd) {  // 512 routed / 1024 shared
  const int tid  = threadIdx.x;
  const int lane = tid & 31;
  const int wv   = tid >> 5;
  const int cg    = blockIdx.x;       // 64-col group of H
  const int rbase = blockIdx.y * 32;
  const int e     = ROUTED ? blockIdx.z : 0;

  int cnt = kT;
  if (ROUTED) {
    cnt = ecount[e]; if (cnt > kCAP) cnt = kCAP;
    if (rbase >= cnt) return;
  }
  const float* Bp = B + (ROUTED ? (size_t)e * kF * kH : (size_t)0);
  const __bf16* Ap = ACT + (ROUTED ? (size_t)e * kCAP * Kd : (size_t)0);

  __shared__ int    s_tok[32];
  __shared__ float  s_wgt[32];
  __shared__ __bf16 sA[32 * 32];   // [row][k]  (TDM destination)
  __shared__ __bf16 sB[64 * 32];   // [col][k] transposed

  if (ROUTED && tid < 32) {
    s_tok[tid] = slot_token[e * kCAP + rbase + tid];
    s_wgt[tid] = slot_weight[e * kCAP + rbase + tid];
  }
  __syncthreads();

  v8f acc0 = {0.f,0.f,0.f,0.f,0.f,0.f,0.f,0.f};
  v8f acc1 = {0.f,0.f,0.f,0.f,0.f,0.f,0.f,0.f};
  const int khalf = lane >> 4;
  const int mrow  = lane & 15;
  const unsigned lds_a = (unsigned)(uintptr_t)&sA[0];

  for (int k0 = 0; k0 < Kd; k0 += 32) {
    // ---- stage A via Tensor Data Mover (bf16 tile, 32 rows x 32 cols) ----
    if (wv == 0) {
      tdm_load_2d_bf16(lds_a, Ap + (size_t)rbase * Kd + k0, 32, 32, Kd);
    }
    // ---- stage B transposed: 64 cols x 32 k (one chunk per thread) ----
    {
      int kp = tid & 15;
      int cc = tid >> 4;              // 0..7
      int lcol0 = cc << 3;
      const float* bp = Bp + (size_t)(k0 + kp * 2) * kH + cg * 64 + lcol0;
      v4f r0a = *(const v4f*)bp;
      v4f r0b = *(const v4f*)(bp + 4);
      v4f r1a = *(const v4f*)(bp + kH);
      v4f r1b = *(const v4f*)(bp + kH + 4);
      for (int j = 0; j < 4; ++j) {
        v2bf p0; p0[0] = (__bf16)r0a[j]; p0[1] = (__bf16)r1a[j];
        *(v2bf*)&sB[(lcol0 + j) * 32 + kp * 2] = p0;
        v2bf p1; p1[0] = (__bf16)r0b[j]; p1[1] = (__bf16)r1b[j];
        *(v2bf*)&sB[(lcol0 + 4 + j) * 32 + kp * 2] = p1;
      }
      if (k0 + 32 < Kd)
        __builtin_prefetch(bp + (size_t)32 * kH, 0, 3);
    }
    if (wv == 0) __builtin_amdgcn_s_wait_tensorcnt(0);  // A tile landed in LDS
    __syncthreads();

    v16bf af0 = ldsFrag(&sA[mrow * 32], khalf);
    v16bf af1 = ldsFrag(&sA[(16 + mrow) * 32], khalf);
    v16bf bf  = ldsFrag(&sB[(wv * 16 + mrow) * 32], khalf);
    acc0 = __builtin_amdgcn_wmma_f32_16x16x32_bf16(false, af0, false, bf,
                                                   (short)0, acc0, false, false);
    acc1 = __builtin_amdgcn_wmma_f32_16x16x32_bf16(false, af1, false, bf,
                                                   (short)0, acc1, false, false);
    __syncthreads();
  }

  const int n   = lane & 15;
  const int mhi = (lane >> 4) << 3;
  const int col = cg * 64 + wv * 16 + n;
  for (int r = 0; r < 8; ++r) {
    int m = r + mhi;
    if (ROUTED) {
      if (rbase + m < cnt)
        atomicAdd(&OUT[(size_t)s_tok[m] * kH + col], acc0[r] * s_wgt[m]);
      if (rbase + 16 + m < cnt)
        atomicAdd(&OUT[(size_t)s_tok[16 + m] * kH + col], acc1[r] * s_wgt[16 + m]);
    } else {
      OUT[(size_t)(rbase + m) * kH + col]      = acc0[r];
      OUT[(size_t)(rbase + 16 + m) * kH + col] = acc1[r];
    }
  }
}

}  // namespace

// ---------------------------------------------------------------------------
// host-side launcher
// ---------------------------------------------------------------------------
extern "C" void kernel_launch(void* const* d_in, const int* in_sizes, int n_in,
                              void* d_out, int out_size, void* d_ws, size_t ws_size,
                              hipStream_t stream) {
  const float* x     = (const float*)d_in[0];  // [T,H]
  const float* gw    = (const float*)d_in[1];  // [E,H]
  const float* bias  = (const float*)d_in[2];  // [E]
  const float* w13   = (const float*)d_in[3];  // [E,H,2F]
  const float* w2    = (const float*)d_in[4];  // [E,F,H]
  const float* sw13  = (const float*)d_in[5];  // [H,2*F*NSH]
  const float* sw2   = (const float*)d_in[6];  // [F*NSH,H]
  float* out = (float*)d_out;                  // [T,H]

  // workspace layout
  char* p = (char*)d_ws;
  int* ecount = (int*)p;                 p += 256;
  int* slot_token = (int*)p;             p += (size_t)kE * kCAP * 4;
  float* slot_weight = (float*)p;        p += (size_t)kE * kCAP * 4;
  __bf16* act = (__bf16*)p;              p += (size_t)kE * kCAP * kF * 2;  // 24 MB
  __bf16* act_s = (__bf16*)p;            p += (size_t)kT * kFS * 2;        // 2 MB

  // 0) init routing state
  moe_init_kernel<<<dim3((kE * kCAP + 255) / 256), 256, 0, stream>>>(
      ecount, slot_token, slot_weight);

  // 1) gate + grouped top-k + dispatch
  gate_topk_kernel<<<dim3(kT), 32, 0, stream>>>(
      x, gw, bias, ecount, slot_token, slot_weight);

  // 2) shared-expert gate/up  (32-row tiles, 64-col groups)
  moe_gateup_kernel<false><<<dim3(kFS / 64, kT / 32), 128, 0, stream>>>(
      x, sw13, nullptr, nullptr, act_s, kFS);

  // 3) routed gate/up (per expert; skips empty row tiles)
  moe_gateup_kernel<true><<<dim3(kF / 64, kCAP / 32, kE), 128, 0, stream>>>(
      x, w13, slot_token, ecount, act, kF);

  // 4) shared-expert down: writes the full output (routed adds on top)
  moe_down_kernel<false><<<dim3(kH / 64, kT / 32), 128, 0, stream>>>(
      act_s, sw2, nullptr, nullptr, nullptr, out, kFS);

  // 5) routed down: scatter-add with routing weights
  moe_down_kernel<true><<<dim3(kH / 64, kCAP / 32, kE), 128, 0, stream>>>(
      act, w2, slot_token, slot_weight, ecount, out, kF);
}
